// LorentzGate_68289980007141
// MI455X (gfx1250) — matrix operand order repaired and checked by reading, not compiled, not run on hardware
//
#include <hip/hip_runtime.h>
#include <hip/hip_bf16.h>

#define DIM   7168
#define NEXP  256
#define NGRP  8
#define GSZ   32          // experts per group
#define TOPG  4
#define TOPK  8
#define TTOK  16384
#define BLK_M 32          // tokens per workgroup
#define KSTEP 32          // K per WMMA / per pipeline stage
#define APAD  40          // LDS row stride in bf16 (80B: 16 distinct banks for 16 rows)

typedef __bf16 v16bf __attribute__((ext_vector_type(16)));
typedef float  v8f   __attribute__((ext_vector_type(8)));
typedef int    v4i   __attribute__((vector_size(16)));       // builtin param 0 pointee
typedef __attribute__((address_space(3))) v4i lds_v4i;       // builtin param 1 pointee

union AB16 { v16bf v; float4 q[2]; };

#if __has_builtin(__builtin_amdgcn_global_load_async_to_lds_b128) && \
    __has_builtin(__builtin_amdgcn_s_wait_asynccnt)
#define USE_ASYNC_LDS 1
#else
#define USE_ASYNC_LDS 0
#endif

// ---------------------------------------------------------------------------
// One-shot weight conversion f32 -> bf16 into workspace (3.67 MB, L2-resident).
// ---------------------------------------------------------------------------
__global__ void cvt_weight_bf16(const float* __restrict__ w,
                                unsigned long long* __restrict__ o, int n4) {
    int i = blockIdx.x * blockDim.x + threadIdx.x;
    if (i >= n4) return;
    float4 f = reinterpret_cast<const float4*>(w)[i];
    union { __bf16 b[4]; unsigned long long u; } p;
    p.b[0] = (__bf16)f.x; p.b[1] = (__bf16)f.y;
    p.b[2] = (__bf16)f.z; p.b[3] = (__bf16)f.w;
    o[i] = p.u;
}

// ---------------------------------------------------------------------------
// Stage one 256x32 bf16 weight slab into LDS (4 x b128 per thread).
// Async path: GLOBAL_LOAD_ASYNC_TO_LDS_B128 (ASYNCcnt), no VGPR round-trip.
// ---------------------------------------------------------------------------
__device__ __forceinline__ void stage_B(const __bf16* __restrict__ wb,
                                        __bf16 (*lB)[APAD], int k, int tid) {
#pragma unroll
    for (int i = 0; i < 4; ++i) {
        const int id = i * 256 + tid;      // 1024 x 16B chunks
        const int e  = id >> 2;            // expert row
        const int kp = (id & 3) * 8;       // bf16 offset within 32-K slab
        const __bf16* gp = wb + (size_t)e * DIM + k + kp;
        __bf16*       lp = &lB[e][kp];
#if USE_ASYNC_LDS
        __builtin_amdgcn_global_load_async_to_lds_b128(
            (v4i*)gp, (lds_v4i*)lp, 0, 0);
#else
        *reinterpret_cast<float4*>(lp) = *reinterpret_cast<const float4*>(gp);
#endif
    }
}

// Stage one 32x32 x-slab: f32 load -> bf16 convert -> ds_store_b64.
__device__ __forceinline__ void stage_A(const float* __restrict__ x,
                                        __bf16 (*lA)[APAD], int m0, int k, int tid) {
    const int r = tid >> 3;                // token row 0..31
    const int c = (tid & 7) * 4;           // col 0..28
    float4 f = *reinterpret_cast<const float4*>(x + (size_t)(m0 + r) * DIM + k + c);
    union { __bf16 b[4]; unsigned long long u; } p;
    p.b[0] = (__bf16)f.x; p.b[1] = (__bf16)f.y;
    p.b[2] = (__bf16)f.z; p.b[3] = (__bf16)f.w;
    *reinterpret_cast<unsigned long long*>(&lA[r][c]) = p.u;
}

// ---------------------------------------------------------------------------
// Fused scoring GEMM (bf16 WMMA, fp32 accum) + sigmoid + grouped top-k route.
// 256 threads = 8 wave32; 32 tokens x 256 experts per block.
//   wave w: M-tile = w>>2 (0..1), N-tiles (w&3)*4 .. +3 (16 experts each)
// ---------------------------------------------------------------------------
__global__ void __launch_bounds__(256)
moe_gate_kernel(const float* __restrict__ x,
                const __bf16* __restrict__ wb,
                const float* __restrict__ bias,
                float* __restrict__ outw,
                int* __restrict__ outi) {
    union Smem {                                    // GEMM buffers alias score matrix
        struct {
            __bf16 A[2][BLK_M][APAD];               // 5 KB  (double-buffered x slab)
            __bf16 B[2][NEXP][APAD];                // 41 KB (double-buffered weight slab)
        } g;
        float S[BLK_M][NEXP + 1];                   // 33 KB (stride 257: conflict-free)
    };
    __shared__ Smem sm;
    __shared__ float lBias[NEXP];

    const int tid  = threadIdx.x;
    const int wave = tid >> 5;
    const int lane = tid & 31;
    const int m0   = blockIdx.x * BLK_M;

    if (tid < NEXP) lBias[tid] = bias[tid];

    const int mt  = wave >> 2;                      // 0..1 : M-tile
    const int ntb = (wave & 3) * 4;                 // first of 4 N-tiles
    const int lhi = lane >> 4;                      // half-wave
    const int llo = lane & 15;

    v8f acc[4] = {};

    // ---- pipeline prologue: fill buffer 0
    stage_B(wb, sm.g.B[0], 0, tid);
    stage_A(x, sm.g.A[0], m0, 0, tid);
#if USE_ASYNC_LDS
    __builtin_amdgcn_s_wait_asynccnt(0);
#endif
    __syncthreads();

    const int kIters = DIM / KSTEP;                 // 224
#pragma unroll 1                                    // single body: acc stays in-place,
    for (int it = 0; it < kIters; ++it) {           // no unroll-rotation register copies
        const int buf = it & 1;
        const int k   = it * KSTEP;

        // ---- issue next stage's global->LDS traffic before computing
        if (it + 1 < kIters) {
            stage_B(wb, sm.g.B[buf ^ 1], k + KSTEP, tid);
            stage_A(x, sm.g.A[buf ^ 1], m0, k + KSTEP, tid);
        }

        // ---- gather all operands (one ds clause), then 4 back-to-back WMMAs
        // A (16x32 bf16, ISA 7.12.2): lane M=llo, K = lhi*8+{0..7} and +16
        AB16 A;
        const __bf16* ap = &sm.g.A[buf][mt * 16 + llo][lhi * 8];
        A.q[0] = *reinterpret_cast<const float4*>(ap);
        A.q[1] = *reinterpret_cast<const float4*>(ap + 16);

        AB16 Bq[4];
#pragma unroll
        for (int j = 0; j < 4; ++j) {
            // B (32x16 bf16): lane N=llo, K = lhi*8+{0..7} and +16
            const __bf16* bp = &sm.g.B[buf][(ntb + j) * 16 + llo][lhi * 8];
            Bq[j].q[0] = *reinterpret_cast<const float4*>(bp);
            Bq[j].q[1] = *reinterpret_cast<const float4*>(bp + 16);
        }
#pragma unroll
        for (int j = 0; j < 4; ++j)
            acc[j] = __builtin_amdgcn_wmma_f32_16x16x32_bf16(
                false, A.v, false, Bq[j].v, (short)0, acc[j], false, false);

#if USE_ASYNC_LDS
        __builtin_amdgcn_s_wait_asynccnt(0);        // staged slab landed in LDS
#endif
        __syncthreads();                            // swap buffers
    }

    // ---- sigmoid epilogue into score matrix (aliases dead GEMM buffers)
#pragma unroll
    for (int j = 0; j < 4; ++j)
#pragma unroll
        for (int r = 0; r < 8; ++r) {
            int mm = mt * 16 + lhi * 8 + r;         // C layout: VGPR r -> M=r (+8 hi half)
            int nn = (ntb + j) * 16 + llo;
            sm.S[mm][nn] = 1.0f / (1.0f + __expf(-acc[j][r]));
        }
    __syncthreads();

    // ---- routing: one thread per token (cost << GEMM)
    if (tid < BLK_M) {
        float* row = sm.S[tid];
        const float NEG = -__builtin_inff();

        // group score = sum of top-2 of (sigmoid + bias) per 32-expert group
        float gs[NGRP];
#pragma unroll
        for (int g = 0; g < NGRP; ++g) {
            float m1 = NEG, m2 = NEG;
            for (int i = 0; i < GSZ; ++i) {
                float v = row[g * GSZ + i] + lBias[g * GSZ + i];
                if (v > m1) { m2 = m1; m1 = v; }
                else if (v > m2) { m2 = v; }
            }
            gs[g] = m1 + m2;
        }
        // top-4 groups (strict '>' => lowest index wins ties, matches lax.top_k)
        bool keep[NGRP] = {};
        for (int it = 0; it < TOPG; ++it) {
            int bg = 0; float bv = NEG;
            for (int g = 0; g < NGRP; ++g)
                if (!keep[g] && gs[g] > bv) { bv = gs[g]; bg = g; }
            keep[bg] = true;
        }
        // top-8 experts among kept groups; mark chosen with -inf in LDS
        int   idx[TOPK];
        float wv[TOPK];
        float wsum = 0.f;
        for (int it = 0; it < TOPK; ++it) {
            int be = 0; float bv = NEG;
            for (int g = 0; g < NGRP; ++g) {
                if (!keep[g]) continue;
                for (int i = 0; i < GSZ; ++i) {
                    int e = g * GSZ + i;
                    float v = row[e] + lBias[e];
                    if (v > bv) { bv = v; be = e; }
                }
            }
            idx[it]  = be;
            wv[it]   = row[be];                     // gather ORIGINAL sigmoid score
            wsum    += wv[it];
            row[be]  = NEG;                         // exclude from later rounds
        }
        const float scale = 2.5f / wsum;            // ROUTE_SCALE / sum
        const int t = m0 + tid;
#pragma unroll
        for (int i = 0; i < TOPK; ++i) {
            outw[t * TOPK + i] = wv[i] * scale;
            outi[t * TOPK + i] = idx[i];
        }
    }
}

// ---------------------------------------------------------------------------
extern "C" void kernel_launch(void* const* d_in, const int* in_sizes, int n_in,
                              void* d_out, int out_size, void* d_ws, size_t ws_size,
                              hipStream_t stream) {
    const float* x    = (const float*)d_in[0];
    const float* w    = (const float*)d_in[1];
    const float* bias = (const float*)d_in[2];

    __bf16* wb = (__bf16*)d_ws;                       // 256*7168*2 = 3.67 MB
    const int n4 = (NEXP * DIM) / 4;
    cvt_weight_bf16<<<(n4 + 255) / 256, 256, 0, stream>>>(
        w, (unsigned long long*)d_ws, n4);

    float* outw = (float*)d_out;                      // [T,8] f32 weights
    int*   outi = (int*)((float*)d_out + (size_t)TTOK * TOPK); // [T,8] i32 indices

    moe_gate_kernel<<<TTOK / BLK_M, 256, 0, stream>>>(x, wb, bias, outw, outi);
}
